// AttnBlock_2920577761574
// MI455X (gfx1250) — matrix operand order repaired and checked by reading, not compiled, not run on hardware
//
#include <hip/hip_runtime.h>

// ---------------------------------------------------------------------------
// Problem sizes (fixed by the reference)
// ---------------------------------------------------------------------------
constexpr int NB = 4;      // batch
constexpr int NT = 4096;   // sequence length
constexpr int ND = 512;    // model dim
constexpr float LN_EPS = 1e-5f;

// Workspace layout (bytes)
constexpr size_t OFF_SIGQ  = 0;
constexpr size_t OFF_SIGK  = 4096;
constexpr size_t OFF_VGATE = 8192;
constexpr size_t OFF_WQBF  = 16384;                       // 512*512 bf16 = 512KB
constexpr size_t OFF_QIN   = OFF_WQBF + 524288;           // B*T*D bf16 = 16MB
constexpr size_t OFF_KBF   = OFF_QIN  + 16777216;         // 16MB
constexpr size_t OFF_VT    = OFF_KBF  + 16777216;         // 16MB (transposed [B][D][T])
constexpr size_t OFF_QBF   = OFF_VT   + 16777216;         // 16MB (post LN/gate Q)

typedef __attribute__((ext_vector_type(16))) __bf16 bf16x16;
typedef __attribute__((ext_vector_type(8)))  __bf16 bf16x8;
typedef __attribute__((ext_vector_type(8)))  float  f32x8;
typedef __attribute__((ext_vector_type(4)))  int    v4i;

// CDNA5 async global->LDS copy path (guarded: falls back to plain copies)
#if defined(__has_builtin)
#if __has_builtin(__builtin_amdgcn_global_load_async_to_lds_b128)
#define HAVE_ASYNC_LDS 1
#endif
#if __has_builtin(__builtin_amdgcn_s_wait_asynccnt)
#define HAVE_WAIT_ASYNC 1
#endif
#endif

__device__ __forceinline__ void async_copy16(const __bf16* gsrc, __bf16* lds) {
#if defined(HAVE_ASYNC_LDS)
    // ROCm clang signature: (v4i* global_src, v4i* lds_dst, imm offset, imm cpol)
    __builtin_amdgcn_global_load_async_to_lds_b128(
        (v4i*)(void*)gsrc, (v4i*)(void*)lds, 0, 0);
#else
    *(bf16x8*)lds = *(const bf16x8*)gsrc;
#endif
}

__device__ __forceinline__ void async_wait0() {
#if defined(HAVE_ASYNC_LDS)
#if defined(HAVE_WAIT_ASYNC)
    __builtin_amdgcn_s_wait_asynccnt(0);
#else
    asm volatile("s_wait_asynccnt 0" ::: "memory");
#endif
#endif
}

__device__ __forceinline__ f32x8 wmma_bf16(bf16x16 a, bf16x16 b, f32x8 c) {
    return __builtin_amdgcn_wmma_f32_16x16x32_bf16(false, a, false, b,
                                                   (short)0, c, false, false);
}

__device__ __forceinline__ bf16x16 make_frag(bf16x8 lo, bf16x8 hi) {
    bf16x16 r;
#pragma unroll
    for (int i = 0; i < 8; ++i) { r[i] = lo[i]; r[i + 8] = hi[i]; }
    return r;
}

__device__ __forceinline__ float sigm(float x) { return 1.0f / (1.0f + __expf(-x)); }

// ---------------------------------------------------------------------------
// Kernel 0: gate vectors
// ---------------------------------------------------------------------------
__global__ __launch_bounds__(512) void gates_kernel(
    const float* __restrict__ qv, const float* __restrict__ kv,
    const float* __restrict__ vv, const float* __restrict__ Ws,
    const float* __restrict__ bs, const float* __restrict__ Wt,
    const float* __restrict__ bt,
    float* __restrict__ sigq, float* __restrict__ sigk, float* __restrict__ vgate)
{
    __shared__ float sv[ND];
    const int t = threadIdx.x;
    sigq[t] = sigm(qv[t]);
    sigk[t] = sigm(kv[t]);
    sv[t]   = sigm(vv[t]);
    __syncthreads();
    float accS = bs[t], accT = bt[t];
    const float* wsRow = Ws + (size_t)t * ND;
    const float* wtRow = Wt + (size_t)t * ND;
    for (int k = 0; k < ND; ++k) {
        float x = sv[k];
        accS += wsRow[k] * x;
        accT += wtRow[k] * x;
    }
    vgate[t] = sigm(accS) * tanhf(accT);
}

// ---------------------------------------------------------------------------
// Kernel 1: elementwise prep -> bf16 operand arrays
// ---------------------------------------------------------------------------
__global__ __launch_bounds__(256) void prep_kernel(
    const float* __restrict__ query, const float* __restrict__ value,
    const float* __restrict__ Wq,
    const float* __restrict__ sigk, const float* __restrict__ vgate,
    __bf16* __restrict__ Qin, __bf16* __restrict__ Wqbf,
    __bf16* __restrict__ Kbf, __bf16* __restrict__ Vt)
{
    const size_t NTOT = (size_t)NB * NT * ND;
    for (size_t idx = (size_t)blockIdx.x * blockDim.x + threadIdx.x;
         idx < NTOT; idx += (size_t)gridDim.x * blockDim.x) {
        int d = (int)(idx & (ND - 1));
        size_t bt = idx >> 9;
        int t = (int)(bt & (NT - 1));
        int b = (int)(bt >> 12);
        Qin[idx] = (__bf16)query[idx];
        float v  = value[idx];
        Kbf[idx] = (__bf16)(v * sigk[d]);
        Vt[((size_t)(b * ND + d)) * NT + t] = (__bf16)(v * vgate[d]);
        if (idx < (size_t)ND * ND) Wqbf[idx] = (__bf16)Wq[idx];
    }
}

// ---------------------------------------------------------------------------
// Kernel 2: Q projection (WMMA bf16) + LayerNorm + sigmoid gate -> Qbf
// ---------------------------------------------------------------------------
__global__ __launch_bounds__(256) void proj_kernel(
    const __bf16* __restrict__ Qin, const __bf16* __restrict__ Wqbf,
    const float* __restrict__ bq, const float* __restrict__ gamma,
    const float* __restrict__ beta, const float* __restrict__ sigq,
    __bf16* __restrict__ Qout)
{
    const int tid = threadIdx.x;
    const int w = tid >> 5, L = tid & 31;
    const int lLo = L & 15, lHi = L >> 4;
    const int miW = w & 3;
    const int nBase = (w >> 2) * 256;
    const int rBase = blockIdx.x * 64;

    f32x8 acc[16];
#pragma unroll
    for (int t = 0; t < 16; ++t)
#pragma unroll
        for (int r = 0; r < 8; ++r) acc[t][r] = 0.0f;

    const __bf16* aRow = Qin + (size_t)(rBase + miW * 16 + lLo) * ND;
    const __bf16* bBase = Wqbf + (size_t)(nBase + lLo) * ND + (lHi << 4);

    bf16x8 alo = *(const bf16x8*)(aRow + (lHi << 4));
    bf16x8 ahi = *(const bf16x8*)(aRow + (lHi << 4) + 8);
    bf16x16 bF = *(const bf16x16*)(bBase);

    for (int kc = 0; kc < 16; ++kc) {
        bf16x16 aF = make_frag(alo, ahi);
        if (kc < 15) {
            int k0 = (kc + 1) * 32 + (lHi << 4);
            alo = *(const bf16x8*)(aRow + k0);
            ahi = *(const bf16x8*)(aRow + k0 + 8);
        }
#pragma unroll
        for (int ni = 0; ni < 16; ++ni) {
            bf16x16 nb = bF;
            if (ni < 15)
                nb = *(const bf16x16*)(bBase + (size_t)(ni + 1) * 16 * ND + kc * 32);
            else if (kc < 15)
                nb = *(const bf16x16*)(bBase + (kc + 1) * 32);
            acc[ni] = wmma_bf16(aF, bF, acc[ni]);
            bF = nb;
        }
    }

    // bias
#pragma unroll
    for (int ni = 0; ni < 16; ++ni) {
        float bb = bq[nBase + ni * 16 + lLo];
#pragma unroll
        for (int r = 0; r < 8; ++r) acc[ni][r] += bb;
    }

    // LayerNorm statistics via LDS float atomics
    __shared__ float sSum[64], sSq[64], sMu[64], sRs[64];
    if (tid < 64) { sSum[tid] = 0.0f; sSq[tid] = 0.0f; }
    __syncthreads();
    float ps[8], pq[8];
#pragma unroll
    for (int r = 0; r < 8; ++r) { ps[r] = 0.0f; pq[r] = 0.0f; }
#pragma unroll
    for (int ni = 0; ni < 16; ++ni)
#pragma unroll
        for (int r = 0; r < 8; ++r) { float v = acc[ni][r]; ps[r] += v; pq[r] += v * v; }
#pragma unroll
    for (int r = 0; r < 8; ++r) {
        int row = miW * 16 + (lHi << 3) + r;
        atomicAdd(&sSum[row], ps[r]);
        atomicAdd(&sSq[row], pq[r]);
    }
    __syncthreads();
    if (tid < 64) {
        float mu = sSum[tid] * (1.0f / ND);
        float var = sSq[tid] * (1.0f / ND) - mu * mu;
        sMu[tid] = mu;
        sRs[tid] = rsqrtf(var + LN_EPS);
    }
    __syncthreads();
    float mu8[8], rs8[8];
#pragma unroll
    for (int r = 0; r < 8; ++r) {
        int row = miW * 16 + (lHi << 3) + r;
        mu8[r] = sMu[row]; rs8[r] = sRs[row];
    }
#pragma unroll
    for (int ni = 0; ni < 16; ++ni) {
        int col = nBase + ni * 16 + lLo;
        float sg = sigq[col];
        float g = gamma[col] * sg;
        float be = beta[col] * sg;
#pragma unroll
        for (int r = 0; r < 8; ++r) {
            float v = (acc[ni][r] - mu8[r]) * rs8[r] * g + be;
            size_t row = (size_t)(rBase + miW * 16 + (lHi << 3) + r);
            Qout[row * ND + col] = (__bf16)v;
        }
    }
}

// ---------------------------------------------------------------------------
// Kernel 3: causal flash attention (WMMA bf16, fp32 accum, online softmax)
// Key block BN=32.  K tile staged in LDS via async copies; the staging of
// tile jb+1 is issued right after all Klds reads of tile jb complete, so the
// DMA overlaps softmax + P*V compute.  Softmax is parallelized over all 256
// threads with split-phase partial max / partial sum reductions.
// ---------------------------------------------------------------------------
__global__ __launch_bounds__(256) void flash_kernel(
    const __bf16* __restrict__ Qbf, const __bf16* __restrict__ Kbf,
    const __bf16* __restrict__ Vt, float* __restrict__ out)
{
    const int tid = threadIdx.x;
    const int w = tid >> 5, L = tid & 31;
    const int lLo = L & 15, lHi = L >> 4;
    const int miW = w >> 1;
    const int half = w & 1;
    const int nBase = half * 256;
    const int b = blockIdx.x >> 6;
    const int iTile = blockIdx.x & 63;
    const float SCALE = 0.044194173824159216f;  // 1/sqrt(512)

    __shared__ __bf16 Klds[32 * ND];   // 32 KB staged K tile
    __shared__ float  Slds[64 * 32];   // 8 KB scores
    __shared__ __bf16 Plds[64 * 32];   // 4 KB probabilities
    __shared__ float  mSh[64], lSh[64], aSh[64];
    __shared__ float  pMax[4][64], pSum[4][64];

    if (tid < 64) { mSh[tid] = -INFINITY; lSh[tid] = 0.0f; }

    f32x8 oAcc[16];
#pragma unroll
    for (int t = 0; t < 16; ++t)
#pragma unroll
        for (int r = 0; r < 8; ++r) oAcc[t][r] = 0.0f;

    const size_t qRow0 = (size_t)b * NT + iTile * 64;
    const __bf16* aRow = Qbf + (qRow0 + miW * 16 + lLo) * ND;

    // softmax phase indices: each thread owns (row, 8-column segment)
    const int smRow = tid & 63;
    const int smSeg = tid >> 6;        // 0..3
    const int smC0  = smSeg * 8;

    const int jbMax = 2 * iTile + 1;   // key blocks of 32, up to the diagonal

    // stage tile 0
    {
        const size_t kTile = (size_t)b * NT * ND;
#pragma unroll
        for (int c = 0; c < 8; ++c) {
            int off = (tid + c * 256) * 8;
            async_copy16(Kbf + kTile + off, &Klds[off]);
        }
    }

    for (int jb = 0; jb <= jbMax; ++jb) {
        const int jBase = jb * 32;

        // ---- (1) current K tile ready ------------------------------------
        async_wait0();
        __syncthreads();

        // ---- S = Q K^T : one 16x16 tile per wave (pipelined A loads) ----
        f32x8 sAcc;
#pragma unroll
        for (int r = 0; r < 8; ++r) sAcc[r] = 0.0f;
        const __bf16* kldsRow = &Klds[(half * 16 + lLo) * ND + (lHi << 4)];
        bf16x8 alo = *(const bf16x8*)(aRow + (lHi << 4));
        bf16x8 ahi = *(const bf16x8*)(aRow + (lHi << 4) + 8);
#pragma unroll
        for (int kc = 0; kc < 16; ++kc) {
            bf16x16 aF = make_frag(alo, ahi);
            if (kc < 15) {
                int k0 = (kc + 1) * 32 + (lHi << 4);
                alo = *(const bf16x8*)(aRow + k0);
                ahi = *(const bf16x8*)(aRow + k0 + 8);
            }
            bf16x16 bF = *(const bf16x16*)(kldsRow + kc * 32);
            sAcc = wmma_bf16(aF, bF, sAcc);
        }
#pragma unroll
        for (int r = 0; r < 8; ++r) {
            int row = miW * 16 + (lHi << 3) + r;
            Slds[row * 32 + half * 16 + lLo] = sAcc[r] * SCALE;
        }
        __syncthreads();   // (2) Slds ready; all Klds reads of tile jb done

        // ---- overlap: issue async staging of tile jb+1 -------------------
        if (jb < jbMax) {
            const size_t kTile = ((size_t)b * NT + jBase + 32) * ND;
#pragma unroll
            for (int c = 0; c < 8; ++c) {
                int off = (tid + c * 256) * 8;
                async_copy16(Kbf + kTile + off, &Klds[off]);
            }
        }
        // preload first V fragments so their loads fly during softmax
        const __bf16* vRow = Vt + ((size_t)b * ND + nBase + lLo) * NT
                                + jBase + (lHi << 4);
        bf16x16 vF0 = *(const bf16x16*)(vRow);
        bf16x16 vF1 = *(const bf16x16*)(vRow + (size_t)16 * NT);

        // ---- parallel online softmax -------------------------------------
        const int climit0 = iTile * 64 + smRow - jBase;   // last valid column
        const int climit = climit0 > 31 ? 31 : climit0;
        {   // phase A: partial max over this thread's 8 columns
            float pm = -INFINITY;
            const float* srow = &Slds[smRow * 32];
#pragma unroll
            for (int c = 0; c < 8; ++c) {
                int cc = smC0 + c;
                float v = (cc <= climit) ? srow[cc] : -INFINITY;
                if (v > pm) pm = v;
            }
            pMax[smSeg][smRow] = pm;
        }
        __syncthreads();   // (3)
        if (tid < 64) {    // phase B: merge max, compute rescale factor
            float mOld = mSh[tid];
            float mNew = mOld;
#pragma unroll
            for (int s = 0; s < 4; ++s) {
                float v = pMax[s][tid];
                if (v > mNew) mNew = v;
            }
            aSh[tid] = __expf(mOld - mNew);
            mSh[tid] = mNew;
        }
        __syncthreads();   // (4)
        {   // phase C: exp + write P + partial sums
            float mNew = mSh[smRow];
            const float* srow = &Slds[smRow * 32];
            __bf16* prow = &Plds[smRow * 32];
            float s = 0.0f;
#pragma unroll
            for (int c = 0; c < 8; ++c) {
                int cc = smC0 + c;
                float p = (cc <= climit) ? __expf(srow[cc] - mNew) : 0.0f;
                prow[cc] = (__bf16)p;
                s += p;
            }
            pSum[smSeg][smRow] = s;
        }
        __syncthreads();   // (5)
        if (tid < 64) {    // phase D: merge sums into running l
            lSh[tid] = lSh[tid] * aSh[tid]
                     + pSum[0][tid] + pSum[1][tid] + pSum[2][tid] + pSum[3][tid];
        }

        // ---- rescale accumulators, then O += P * V (pipelined V loads) --
        float al[8];
#pragma unroll
        for (int r = 0; r < 8; ++r) al[r] = aSh[miW * 16 + (lHi << 3) + r];
#pragma unroll
        for (int t = 0; t < 16; ++t)
#pragma unroll
            for (int r = 0; r < 8; ++r) oAcc[t][r] *= al[r];

        const __bf16* pRow = &Plds[(miW * 16 + lLo) * 32];
        bf16x8 plo = *(const bf16x8*)(pRow + (lHi << 4));
        bf16x8 phi = *(const bf16x8*)(pRow + (lHi << 4) + 8);
        bf16x16 aP = make_frag(plo, phi);

        bf16x16 vF = vF0;
        bf16x16 vN = vF1;
#pragma unroll
        for (int t = 0; t < 16; ++t) {
            bf16x16 nv = vN;
            if (t < 14)
                vN = *(const bf16x16*)(vRow + (size_t)(t + 2) * 16 * NT);
            oAcc[t] = wmma_bf16(aP, vF, oAcc[t]);
            vF = nv;
        }
        // loop re-enters at (1): async wait + barrier orders everything
    }
    __syncthreads();       // lSh final values visible to all threads

    // ---- epilogue: divide by l, write fp32 output ------------------------
    float linv[8];
#pragma unroll
    for (int r = 0; r < 8; ++r)
        linv[r] = 1.0f / lSh[miW * 16 + (lHi << 3) + r];
#pragma unroll
    for (int t = 0; t < 16; ++t) {
        int col = nBase + t * 16 + lLo;
#pragma unroll
        for (int r = 0; r < 8; ++r) {
            size_t row = qRow0 + miW * 16 + (lHi << 3) + r;
            out[row * ND + col] = oAcc[t][r] * linv[r];
        }
    }
}

// ---------------------------------------------------------------------------
// Host-side launcher
// ---------------------------------------------------------------------------
extern "C" void kernel_launch(void* const* d_in, const int* in_sizes, int n_in,
                              void* d_out, int out_size, void* d_ws, size_t ws_size,
                              hipStream_t stream) {
    const float* query = (const float*)d_in[0];
    const float* value = (const float*)d_in[1];
    const float* qvec  = (const float*)d_in[2];
    const float* kvec  = (const float*)d_in[3];
    const float* vvec  = (const float*)d_in[4];
    const float* Wq    = (const float*)d_in[5];
    const float* bq    = (const float*)d_in[6];
    const float* lng   = (const float*)d_in[7];
    const float* lnb   = (const float*)d_in[8];
    const float* Ws    = (const float*)d_in[9];
    const float* bs    = (const float*)d_in[10];
    const float* Wt    = (const float*)d_in[11];
    const float* bt    = (const float*)d_in[12];
    float* out = (float*)d_out;

    char* ws = (char*)d_ws;
    float*  sigq  = (float*)(ws + OFF_SIGQ);
    float*  sigk  = (float*)(ws + OFF_SIGK);
    float*  vgate = (float*)(ws + OFF_VGATE);
    __bf16* Wqbf  = (__bf16*)(ws + OFF_WQBF);
    __bf16* Qin   = (__bf16*)(ws + OFF_QIN);
    __bf16* Kbf   = (__bf16*)(ws + OFF_KBF);
    __bf16* Vt    = (__bf16*)(ws + OFF_VT);
    __bf16* Qbf   = (__bf16*)(ws + OFF_QBF);

    gates_kernel<<<1, 512, 0, stream>>>(qvec, kvec, vvec, Ws, bs, Wt, bt,
                                        sigq, sigk, vgate);
    prep_kernel<<<2048, 256, 0, stream>>>(query, value, Wq, sigk, vgate,
                                          Qin, Wqbf, Kbf, Vt);
    proj_kernel<<<(NB * NT) / 64, 256, 0, stream>>>(Qin, Wqbf, bq, lng, lnb,
                                                    sigq, Qbf);
    flash_kernel<<<NB * (NT / 64), 256, 0, stream>>>(Qbf, Kbf, Vt, out);
}
